// GConv_12249246728621
// MI455X (gfx1250) — matrix operand order repaired and checked by reading, not compiled, not run on hardware
//
#include <hip/hip_runtime.h>

#define DIM 64
#define LN_EPS 1e-5f

typedef __attribute__((ext_vector_type(16))) _Float16 v16h;
typedef __attribute__((ext_vector_type(8)))  float    v8f;

// ---------------------------------------------------------------------------
// One wave (32 lanes, EXEC all-ones) accumulates a 16x16 f32 tile:
//   C += (rowscale_per_row * A[m0:m0+16, 0:64]) @ W[n0:n0+16, 0:64]^T
// A row-major [Nrows,64]; W row-major [64,64] (torch Linear: out = x @ W^T).
// Lane layout per CDNA5 ISA 7.12.2 (V_WMMA_F32_16X16X32_F16, wave32):
//   A (16x32 f16): lane L<16 row M=L, halves = K {kk+0..7, kk+16..23};
//                  lane L>=16 same rows, K {kk+8..15, kk+24..31}.
//   B (32x16 f16): lane L holds output column N=L%16 (i.e. W row n0+L%16);
//                  same K split as A: lanes 0-15 {kk+0..7, kk+16..23}, 16-31 {kk+8..15, kk+24..31}.
//   C/D (16x16 f32): VGPR v -> M = v + 8*(L>=16), N = L%16.
// ---------------------------------------------------------------------------
__device__ __forceinline__ v8f wmma_tile_f16(const float* __restrict__ A, int m0,
                                             float rowscale,
                                             const float* __restrict__ W, int n0,
                                             v8f c, int Nrows)
{
  const int lane = threadIdx.x & 31;
  const int half = lane >> 4;
  const int row  = lane & 15;
  const int ar   = (m0 + row < Nrows) ? (m0 + row) : (Nrows - 1);  // clamp reads
  const float* __restrict__ Abase = A + (size_t)ar * DIM;
  const float* __restrict__ Wbase = W + (size_t)(n0 + row) * DIM;

#pragma unroll
  for (int kk = 0; kk < DIM; kk += 32) {
    const float* arow = Abase + kk + half * 8;   // 32B aligned
    float av[16];
    *(float4*)(&av[0])  = *(const float4*)(arow + 0);
    *(float4*)(&av[4])  = *(const float4*)(arow + 4);
    *(float4*)(&av[8])  = *(const float4*)(arow + 16);
    *(float4*)(&av[12]) = *(const float4*)(arow + 20);

    const float* wrow = Wbase + kk + half * 8;   // 32B aligned
    float bv[16];
    *(float4*)(&bv[0])  = *(const float4*)(wrow + 0);
    *(float4*)(&bv[4])  = *(const float4*)(wrow + 4);
    *(float4*)(&bv[8])  = *(const float4*)(wrow + 16);
    *(float4*)(&bv[12]) = *(const float4*)(wrow + 20);

    v16h a, b;
#pragma unroll
    for (int e = 0; e < 16; ++e) a[e] = (_Float16)(av[e] * rowscale);
#pragma unroll
    for (int e = 0; e < 16; ++e) b[e] = (_Float16)bv[e];

    // (neg_a, A, neg_b, B, c_mod, C, reuse_a, reuse_b)
    c = __builtin_amdgcn_wmma_f32_16x16x32_f16(false, a, false, b, (short)0, c,
                                               false, false);
  }
  return c;
}

// Tile store epilogue. The tile-overflow test is wave-uniform (m0 uniform),
// so the common full-tile case is one uniform branch + 8 straight-line
// coalesced b32 stores (each v: a half-wave writes 64B contiguous).
template <bool RELU>
__device__ __forceinline__ void store_tile(float* __restrict__ dst, v8f c,
                                           float bias, int m0, int n0, int Nrows)
{
  const int lane = threadIdx.x & 31;
  const int half = lane >> 4;
  const int row  = lane & 15;
  float* __restrict__ base = dst + (size_t)(m0 + 8 * half) * DIM + n0 + row;

  if (m0 + 16 <= Nrows) {            // fast path: full tile, no per-lane guards
#pragma unroll
    for (int v = 0; v < 8; ++v) {
      float val = c[v] + bias;
      if (RELU) val = fmaxf(val, 0.0f);
      base[(size_t)v * DIM] = val;
    }
  } else {                           // ragged tail tile (not hit when N%16==0)
#pragma unroll
    for (int v = 0; v < 8; ++v) {
      const int m = m0 + v + 8 * half;
      if (m < Nrows) {
        float val = c[v] + bias;
        if (RELU) val = fmaxf(val, 0.0f);
        base[(size_t)v * DIM] = val;
      }
    }
  }
}

// hp = relu(h @ Wp^T + bp)
__global__ void __launch_bounds__(32)
sage_proj_relu(const float* __restrict__ h, const float* __restrict__ Wp,
               const float* __restrict__ bp, float* __restrict__ hp, int Nrows)
{
  const int m0 = blockIdx.x * 16, n0 = blockIdx.y * 16;
  v8f c = {0.f, 0.f, 0.f, 0.f, 0.f, 0.f, 0.f, 0.f};
  c = wmma_tile_f16(h, m0, 1.0f, Wp, n0, c, Nrows);
  const float bias = bp[n0 + (threadIdx.x & 15)];
  store_tile<true>(hp, c, bias, m0, n0, Nrows);
}

// pre = (agg / max(deg,1)) @ Wl^T + bl + h @ Wr^T
__global__ void __launch_bounds__(32)
sage_combine(const float* __restrict__ agg, const float* __restrict__ deg,
             const float* __restrict__ h,
             const float* __restrict__ Wl, const float* __restrict__ Wr,
             const float* __restrict__ bl, float* __restrict__ pre, int Nrows)
{
  const int m0 = blockIdx.x * 16, n0 = blockIdx.y * 16;
  const int row = threadIdx.x & 15;
  const int ar  = (m0 + row < Nrows) ? (m0 + row) : (Nrows - 1);
  const float rdeg = 1.0f / fmaxf(deg[ar], 1.0f);   // mean aggregation, clip>=1

  v8f c = {0.f, 0.f, 0.f, 0.f, 0.f, 0.f, 0.f, 0.f};
  c = wmma_tile_f16(agg, m0, rdeg, Wl, n0, c, Nrows);  // lin_l(mean_agg)
  c = wmma_tile_f16(h,   m0, 1.0f, Wr, n0, c, Nrows);  // + lin_r(x)

  const float bias = bl[n0 + row];
  store_tile<false>(pre, c, bias, m0, n0, Nrows);
}

// Wave-per-edge scatter-add: agg[dst] += hp[src]; deg[dst] += 1.
// hp/agg are L2-resident (12.8 MB each in a 192 MB L2), so these are
// L2-side global_atomic_add_f32 ops, not HBM round trips.
__global__ void __launch_bounds__(256)
sage_scatter(const float* __restrict__ hp, const int* __restrict__ src,
             const int* __restrict__ dst, float* __restrict__ agg,
             float* __restrict__ deg, int E)
{
  const int e = blockIdx.x * 8 + (threadIdx.x >> 5);
  if (e >= E) return;
  const int lane = threadIdx.x & 31;
  const int s = src[e], d = dst[e];
  const float v0 = hp[(size_t)s * DIM + lane];
  const float v1 = hp[(size_t)s * DIM + 32 + lane];
  atomicAdd(&agg[(size_t)d * DIM + lane], v0);
  atomicAdd(&agg[(size_t)d * DIM + 32 + lane], v1);
  if (lane == 0) atomicAdd(&deg[d], 1.0f);
}

// Wave-per-row LayerNorm (biased variance, eps=1e-5), wave32 shuffle reduce.
__global__ void __launch_bounds__(256)
sage_layernorm(const float* __restrict__ pre, const float* __restrict__ g,
               const float* __restrict__ be, float* __restrict__ out, int Nrows)
{
  const int r = blockIdx.x * 8 + (threadIdx.x >> 5);
  if (r >= Nrows) return;
  const int lane = threadIdx.x & 31;
  const float x0 = pre[(size_t)r * DIM + lane];
  const float x1 = pre[(size_t)r * DIM + 32 + lane];
  float s = x0 + x1;
  float q = x0 * x0 + x1 * x1;
#pragma unroll
  for (int off = 16; off > 0; off >>= 1) {
    s += __shfl_xor(s, off, 32);
    q += __shfl_xor(q, off, 32);
  }
  const float mu  = s * (1.0f / DIM);
  const float var = q * (1.0f / DIM) - mu * mu;
  const float rs  = rsqrtf(var + LN_EPS);
  out[(size_t)r * DIM + lane]      = (x0 - mu) * rs * g[lane]      + be[lane];
  out[(size_t)r * DIM + 32 + lane] = (x1 - mu) * rs * g[32 + lane] + be[32 + lane];
}

__global__ void __launch_bounds__(256)
zero_f32(float* __restrict__ p, size_t n)
{
  size_t i = (size_t)blockIdx.x * 256 + threadIdx.x;
  const size_t st = (size_t)gridDim.x * 256;
  for (; i < n; i += st) p[i] = 0.0f;
}

extern "C" void kernel_launch(void* const* d_in, const int* in_sizes, int n_in,
                              void* d_out, int out_size, void* d_ws, size_t ws_size,
                              hipStream_t stream)
{
  const float* x      = (const float*)d_in[0];
  const int*   ei     = (const int*)  d_in[1];  // [2,E]: src row then dst row
  const float* W_proj = (const float*)d_in[2];
  const float* b_proj = (const float*)d_in[3];
  const float* W_l    = (const float*)d_in[4];
  const float* b_l    = (const float*)d_in[5];
  const float* W_r    = (const float*)d_in[6];
  const float* gamma  = (const float*)d_in[7];
  const float* beta   = (const float*)d_in[8];

  const int Nn = in_sizes[0] / DIM;            // 50000
  const int Ee = in_sizes[1] / 2;              // 800000
  const int Ln = in_sizes[2] / (DIM * DIM);    // 2

  // Workspace: agg[N*64] | deg[N] | hp[N*64] (aliased as `pre`) | h1[N*64]
  float* agg = (float*)d_ws;
  float* deg = agg + (size_t)Nn * DIM;
  float* hp  = deg + Nn;
  float* h1  = hp  + (size_t)Nn * DIM;

  const int tilesM = (Nn + 15) / 16;
  const dim3 gemm_grid(tilesM, DIM / 16);

  for (int l = 0; l < Ln; ++l) {
    const float* hin  = (l == 0) ? x : h1;
    float*       hout = (l == Ln - 1) ? (float*)d_out : h1;
    const float* Wp = W_proj + (size_t)l * DIM * DIM;
    const float* bp = b_proj + (size_t)l * DIM;
    const float* Wl = W_l    + (size_t)l * DIM * DIM;
    const float* bl = b_l    + (size_t)l * DIM;
    const float* Wr = W_r    + (size_t)l * DIM * DIM;
    const float* g  = gamma  + (size_t)l * DIM;
    const float* be = beta   + (size_t)l * DIM;

    // agg and deg are contiguous -> one clear of N*(D+1) floats
    zero_f32<<<2048, 256, 0, stream>>>(agg, (size_t)Nn * (DIM + 1));
    sage_proj_relu<<<gemm_grid, 32, 0, stream>>>(hin, Wp, bp, hp, Nn);
    sage_scatter<<<(Ee + 7) / 8, 256, 0, stream>>>(hp, ei, ei + Ee, agg, deg, Ee);
    sage_combine<<<gemm_grid, 32, 0, stream>>>(agg, deg, hin, Wl, Wr, bl,
                                               hp /* pre (aliases hp) */, Nn);
    sage_layernorm<<<(Nn + 7) / 8, 256, 0, stream>>>(hp, g, be, hout, Nn);
  }
}
